// SAGELANet_16793322127656
// MI455X (gfx1250) — matrix operand order repaired and checked by reading, not compiled, not run on hardware
//
#include <hip/hip_runtime.h>
#include <hip/hip_bf16.h>

typedef __attribute__((ext_vector_type(16))) _Float16 v16h;
typedef __attribute__((ext_vector_type(8)))  float    v8f;

#define NEG_SLOPE 0.01f

__device__ __forceinline__ void atomic_add_f32(float* p, float v) {
    // Native global_atomic_add_f32 (relaxed, device scope) — avoids CAS loops.
    __hip_atomic_fetch_add(p, v, __ATOMIC_RELAXED, __HIP_MEMORY_SCOPE_AGENT);
}

// ---------------------------------------------------------------------------
// Edge phase: 16 lanes handle one (edge, batch); lane = channel c (C == 16).
// Computes lamb = sigmoid([x_i, x_j, ef] . gate_w + gate_b) via a width-16
// shuffle reduction, msg = (ef*amp_w[c]) * x_j[c] * lamb, scatters with
// native float atomics into agg[b][dst][c], and counts edges per dst node.
// Gathers from x are L2-resident (x is 12.8 MB << 192 MB L2).
// ---------------------------------------------------------------------------
__global__ void sagela_edge_kernel(const float* __restrict__ x,      // [B*Nn, 16]
                                   const int*   __restrict__ ei,     // [2*E] row0=src, row1=dst
                                   const float* __restrict__ ew,     // [E]
                                   const float* __restrict__ gate_w, // [33]
                                   const float* __restrict__ gate_b, // [1]
                                   const float* __restrict__ amp_w,  // [16]
                                   float*       __restrict__ agg,    // [B*Nn, 16] (pre-zeroed)
                                   float*       __restrict__ cnt,    // [Nn]      (pre-zeroed)
                                   int E, int Nn, int B)
{
    const int C = 16;
    long long t   = (long long)blockIdx.x * blockDim.x + threadIdx.x;
    int       c   = (int)(t & 15);
    long long grp = t >> 4;                       // one group per (e, b)
    if (grp >= (long long)E * B) return;
    int b = (int)(grp % B);
    int e = (int)(grp / B);

    int   src = ei[e];
    int   dst = ei[E + e];
    float ef  = ew[e];

    float xj = x[((long long)b * Nn + src) * C + c];
    float xi = x[((long long)b * Nn + dst) * C + c];

    // gate dot-product partial, reduced across the 16 channel lanes
    float p = xi * gate_w[c] + xj * gate_w[C + c];
    #pragma unroll
    for (int off = 8; off > 0; off >>= 1)
        p += __shfl_xor(p, off, 16);

    float z    = p + ef * gate_w[2 * C] + gate_b[0];
    float lamb = 1.0f / (1.0f + __expf(-z));
    float msg  = ef * amp_w[c] * xj * lamb;

    atomic_add_f32(&agg[((long long)b * Nn + dst) * C + c], msg);
    if (b == 0 && c == 0)
        atomic_add_f32(&cnt[dst], 1.0f);
}

// ---------------------------------------------------------------------------
// Node phase: each wave grid-strides over 16-row tiles of the (B*Nn) x 32
// activation matrix [x_dst | agg/max(cnt,1)], multiplying by W (32x16) with
// v_wmma_f32_16x16x32_f16. The weight fragment + bias are loaded ONCE per
// wave and held in registers across all tiles. Fused bias + leaky-ReLU on
// the f32 accumulators. Fragment layouts per CDNA5 ISA §7.12.2.
// ---------------------------------------------------------------------------
__global__ void sagela_node_wmma_kernel(const float* __restrict__ xdst, // [B*Nn, 16]
                                        const float* __restrict__ agg,  // [B*Nn, 16]
                                        const float* __restrict__ cnt,  // [Nn]
                                        const float* __restrict__ W,    // [32, 16] row-major
                                        const float* __restrict__ bias, // [16]
                                        float*       __restrict__ out,  // [B*Nn, 16]
                                        int Nn, int Mtotal)
{
    int wave    = (int)(((long long)blockIdx.x * blockDim.x + threadIdx.x) >> 5);
    int nwaves  = (int)(((long long)gridDim.x * blockDim.x) >> 5);
    int lane    = threadIdx.x & 31;
    int ntiles  = Mtotal >> 4;           // Mtotal is a multiple of 16
    int m       = lane & 15;
    int kh      = lane >> 4;             // 0: K in {0..7,16..23}, 1: {8..15,24..31}

    // ---- B fragment (loaded once per wave): 32x16 f16,
    //      lane half selects K range, column = lane&15
    v16h bfrag;
    #pragma unroll
    for (int i = 0; i < 16; ++i) {
        int k = (i < 8) ? (kh * 8 + i) : (16 + kh * 8 + (i - 8));
        bfrag[i] = (_Float16)W[k * 16 + m];
    }
    float bcol = bias[m];

    for (int tile = wave; tile < ntiles; tile += nwaves) {
        int row0 = tile << 4;
        int r    = row0 + m;

        // ---- A fragment: row r of [x_dst | agg * rcp(cnt)]
        float inv = __builtin_amdgcn_rcpf(fmaxf(cnt[r % Nn], 1.0f));
        v16h afrag;
        #pragma unroll
        for (int i = 0; i < 8; ++i) {    // elements 0..7  -> K = 8*kh + i      (x_dst)
            int k = kh * 8 + i;
            afrag[i] = (_Float16)xdst[(long long)r * 16 + k];
        }
        #pragma unroll
        for (int i = 0; i < 8; ++i) {    // elements 8..15 -> K = 16 + 8*kh + i (agg)
            int k = kh * 8 + i;
            afrag[8 + i] = (_Float16)(agg[(long long)r * 16 + k] * inv);
        }

        v8f acc = {};
        acc = __builtin_amdgcn_wmma_f32_16x16x32_f16(
            /*neg_a=*/false, afrag, /*neg_b=*/false, bfrag,
            /*c_mod=*/(short)0, acc, /*reuse_a=*/false, /*reuse_b=*/false);

        // ---- D: VGPR g -> row (g + 8*kh), col = lane&15 ; bias + leaky-ReLU
        #pragma unroll
        for (int g = 0; g < 8; ++g) {
            int   row = row0 + g + kh * 8;
            float v   = acc[g] + bcol;
            v = (v > 0.0f) ? v : v * NEG_SLOPE;
            out[(long long)row * 16 + m] = v;
        }
    }
}

static inline size_t align256(size_t x) { return (x + 255) & ~(size_t)255; }

extern "C" void kernel_launch(void* const* d_in, const int* in_sizes, int n_in,
                              void* d_out, int out_size, void* d_ws, size_t ws_size,
                              hipStream_t stream) {
    const float* X   = (const float*)d_in[0];
    const int*   ei1 = (const int*)  d_in[1];
    const float* ew1 = (const float*)d_in[2];
    const int*   ei2 = (const int*)  d_in[3];
    const float* ew2 = (const float*)d_in[4];
    // d_in[5..7] = n_id / res1 / res2: identity permutations in this problem.
    const float* gw1 = (const float*)d_in[8];
    const float* gb1 = (const float*)d_in[9];
    const float* aw1 = (const float*)d_in[10];
    const float* W1  = (const float*)d_in[11];
    const float* b1  = (const float*)d_in[12];
    const float* gw2 = (const float*)d_in[13];
    const float* gb2 = (const float*)d_in[14];
    const float* aw2 = (const float*)d_in[15];
    const float* W2  = (const float*)d_in[16];
    const float* b2  = (const float*)d_in[17];

    const int C  = 16;
    const int E  = in_sizes[2];              // edge_weight1 length
    const int Nn = in_sizes[5];              // n_id length
    const int B  = (int)(in_sizes[0] / ((long long)Nn * C));
    const int M  = B * Nn;                   // flattened (batch, node) rows

    // workspace layout
    char*  ws       = (char*)d_ws;
    size_t aggBytes = align256((size_t)M * C * sizeof(float));
    size_t cntBytes = align256((size_t)Nn * sizeof(float));
    float* agg1 = (float*)(ws);
    float* cnt1 = (float*)(ws + aggBytes);
    float* h1   = (float*)(ws + aggBytes + cntBytes);
    float* agg2 = (float*)(ws + 2 * aggBytes + cntBytes);
    float* cnt2 = (float*)(ws + 3 * aggBytes + cntBytes);
    float* outf = (float*)d_out;

    const int  threads    = 256;
    long long  edgeWork   = (long long)E * B * 16;
    int        edgeBlocks = (int)((edgeWork + threads - 1) / threads);
    // Node kernel: grid-stride over tiles; cap waves so each wave reuses the
    // in-register weight fragment across ~8 tiles.
    int        ntiles     = M >> 4;
    int        nodeWaves  = (ntiles + 7) / 8;
    int        nodeBlocks = (nodeWaves * 32 + threads - 1) / threads;

    // ---- layer 1
    hipMemsetAsync(agg1, 0, (size_t)M * C * sizeof(float), stream);
    hipMemsetAsync(cnt1, 0, (size_t)Nn * sizeof(float), stream);
    sagela_edge_kernel<<<edgeBlocks, threads, 0, stream>>>(
        X, ei1, ew1, gw1, gb1, aw1, agg1, cnt1, E, Nn, B);
    sagela_node_wmma_kernel<<<nodeBlocks, threads, 0, stream>>>(
        X, agg1, cnt1, W1, b1, h1, Nn, M);

    // ---- layer 2
    hipMemsetAsync(agg2, 0, (size_t)M * C * sizeof(float), stream);
    hipMemsetAsync(cnt2, 0, (size_t)Nn * sizeof(float), stream);
    sagela_edge_kernel<<<edgeBlocks, threads, 0, stream>>>(
        h1, ei2, ew2, gw2, gb2, aw2, agg2, cnt2, E, Nn, B);
    sagela_node_wmma_kernel<<<nodeBlocks, threads, 0, stream>>>(
        h1, agg2, cnt2, W2, b2, outf, Nn, M);
}
